// VAE_14834817040712
// MI455X (gfx1250) — compile-verified, matrix operand order checked by reading, not run on hardware
//
#include <hip/hip_runtime.h>
#include <hip/hip_bf16.h>
#include <math.h>

typedef __bf16 bf16_t;
typedef __attribute__((ext_vector_type(16))) __bf16 v16bf;
typedef __attribute__((ext_vector_type(8)))  __bf16 v8bf;
typedef __attribute__((ext_vector_type(8)))  float  v8f;
typedef int v4i_t __attribute__((vector_size(4 * sizeof(int))));

#define V_  32000
#define E_  512
#define H_  1024
#define G3H (3 * H_)
#define Z_  256
#define B_  64
#define S_  256
#define T_  64
#define BOS_ID 1
#define EOS_ID 2

#define LDS_PAD 8   // pad A-tile rows by 8 bf16 (16B): row stride % 256B != 0

// ---------------------------------------------------------------------------
// Async global->LDS staging (CDNA5 GLOBAL_LOAD_ASYNC_TO_LDS_B128, ASYNCcnt),
// with a synchronous fallback when the builtin is absent on this toolchain.
// Builtin signature (from clang diagnostic):
//   void(v4i addrspace(1)*, v4i addrspace(3)*, Imm int offset, Imm int cpol)
// ---------------------------------------------------------------------------
#if __has_builtin(__builtin_amdgcn_global_load_async_to_lds_b128)
#define HAVE_ASYNC_LDS 1
__device__ __forceinline__ void cp16_g2l(const bf16_t* g, bf16_t* l) {
    __builtin_amdgcn_global_load_async_to_lds_b128(
        (__attribute__((address_space(1))) v4i_t*)const_cast<bf16_t*>(g),
        (__attribute__((address_space(3))) v4i_t*)l,
        0, 0);
}
__device__ __forceinline__ void wait_async_done() {
    asm volatile("s_wait_asynccnt 0x0" ::: "memory");
}
#else
#define HAVE_ASYNC_LDS 0
__device__ __forceinline__ void cp16_g2l(const bf16_t* g, bf16_t* l) {
    *reinterpret_cast<v8bf*>(l) = *reinterpret_cast<const v8bf*>(g);
}
__device__ __forceinline__ void wait_async_done() {}
#endif

// ---------------------------------------------------------------------------
// WMMA operand load: 16 bf16 per lane as two contiguous 16-byte chunks.
// A-matrix 16x32 bf16 lane layout (ISA 7.12.2): lane half selects K-subrange
// {half*8 .. half*8+7} for elements 0..7 and {16+half*8 ..} for elements 8..15.
// Works for both global (global_load_b128) and LDS (ds_load_b128) pointers.
// ---------------------------------------------------------------------------
__device__ __forceinline__ v16bf load_op16(const bf16_t* p) {
    v8bf lo = *reinterpret_cast<const v8bf*>(p);
    v8bf hi = *reinterpret_cast<const v8bf*>(p + 16);
    v16bf r;
#pragma unroll
    for (int i = 0; i < 8; ++i) { r[i] = lo[i]; r[i + 8] = hi[i]; }
    return r;
}

// ---------------------------------------------------------------------------
// bf16 WMMA GEMM: C[M,N] = A[M,K] * B[N,K]^T + bias[N], fp32 accumulate.
// Block = 256 threads = 8 waves; block tile = 16 (M) x 128 (N); one wave per
// 16x16 D tile. The shared 16xK A tile is staged to LDS once (async-to-LDS),
// so global traffic in the K loop is B only (streams from L2 for W_voc).
// tokens != nullptr gathers A rows from an embedding table instead.
// grid = (N/128, M/16); dynamic LDS = 16*(K+LDS_PAD)*2 bytes.
// ---------------------------------------------------------------------------
__global__ __launch_bounds__(256) void k_wmma_gemm(
    const bf16_t* __restrict__ A, const int* __restrict__ tokens,
    const bf16_t* __restrict__ Bm, const float* __restrict__ bias,
    float* __restrict__ C, int N, int K) {
    extern __shared__ bf16_t As[];          // 16 rows x (K + LDS_PAD)
    const int lds_stride = K + LDS_PAD;

    const int lane = threadIdx.x & 31;
    const int wave = threadIdx.x >> 5;
    const int half = lane >> 4;
    const int r    = lane & 15;
    const int n0 = blockIdx.x * 128 + wave * 16;
    const int m0 = blockIdx.y * 16;

    // ---- stage A tile (16 x K bf16) into LDS, 16B chunks across the block
    const int chunks_per_row = K >> 3;      // 8 bf16 per 16B chunk
    const int total_chunks   = chunks_per_row << 4;
    for (int c = threadIdx.x; c < total_chunks; c += 256) {
        const int row = c / chunks_per_row;
        const int cc  = c - row * chunks_per_row;
        const int src_row = tokens ? tokens[m0 + row] : (m0 + row);
        const bf16_t* src = A + (size_t)src_row * K + cc * 8;
        bf16_t*       dst = As + row * lds_stride + cc * 8;
        cp16_g2l(src, dst);
    }
    wait_async_done();
    __syncthreads();

    // ---- K loop: A from LDS (ds_load_b128), B from global (global_load_b128)
    const bf16_t* a_row = As + r * lds_stride + half * 8;
    const bf16_t* b_row = Bm + (size_t)(n0 + r) * K + half * 8;

    v8f acc = {};
#pragma unroll 4
    for (int kb = 0; kb < K; kb += 32) {
        __builtin_prefetch((const void*)(b_row + kb + 256), 0, 0);
        v16bf a = load_op16(a_row + kb);
        v16bf b = load_op16(b_row + kb);
        acc = __builtin_amdgcn_wmma_f32_16x16x32_bf16(
            false, a, false, b, (short)0, acc, false, false);
    }
    const int n = n0 + r;
    const float bv = bias[n];
#pragma unroll
    for (int v = 0; v < 8; ++v) {
        const int m = m0 + v + 8 * half;
        C[(size_t)m * N + n] = acc[v] + bv;
    }
}

// ---------------------------------------------------------------------------
// GRU gate fusion: r,z,n gates + state update + bf16 mirror for next GEMM.
// ---------------------------------------------------------------------------
__global__ __launch_bounds__(256) void k_gru_gate(
    const float* __restrict__ gx, const float* __restrict__ hg,
    float* __restrict__ h, bf16_t* __restrict__ hb, int bc) {
    const int idx = blockIdx.x * blockDim.x + threadIdx.x;
    if (idx >= B_ * H_) return;
    const int b = idx / H_;
    const int j = idx - b * H_;
    const float* g  = bc ? gx : (gx + (size_t)b * G3H);
    const float* hh = hg + (size_t)b * G3H;
    const float xr = g[j], xz = g[H_ + j], xn = g[2 * H_ + j];
    const float hr = hh[j], hz = hh[H_ + j], hn = hh[2 * H_ + j];
    const float rr = 1.0f / (1.0f + __expf(-(xr + hr)));
    const float zz = 1.0f / (1.0f + __expf(-(xz + hz)));
    const float nn = tanhf(xn + rr * hn);
    const float hnew = (1.0f - zz) * nn + zz * h[idx];
    h[idx]  = hnew;
    hb[idx] = (bf16_t)hnew;
}

// Small fp32 dense: out[M,N] = A[M,K] @ W[N,K]^T + bias, act 0=none 1=softplus
__global__ __launch_bounds__(256) void k_dense_f32(
    const float* __restrict__ A, const float* __restrict__ W,
    const float* __restrict__ bias, float* __restrict__ out,
    int M, int N, int K, int act) {
    const int idx = blockIdx.x * blockDim.x + threadIdx.x;
    if (idx >= M * N) return;
    const int m = idx / N;
    const int n = idx - m * N;
    const float* a = A + (size_t)m * K;
    const float* w = W + (size_t)n * K;
    float acc = 0.0f;
    for (int k = 0; k < K; ++k) acc = fmaf(a[k], w[k], acc);
    acc += bias[n];
    if (act == 1) acc = (acc > 20.0f) ? acc : log1pf(__expf(acc));
    out[idx] = acc;
}

__global__ void k_reparam(const float* __restrict__ loc,
                          const float* __restrict__ scale,
                          const float* __restrict__ eps,
                          float* __restrict__ z,
                          float* __restrict__ out_loc,
                          float* __restrict__ out_scale) {
    const int i = blockIdx.x * blockDim.x + threadIdx.x;
    if (i >= B_ * Z_) return;
    const float l = loc[i], s = scale[i];
    z[i] = l + eps[i] * __expf(0.5f * s);
    out_loc[i]   = l;
    out_scale[i] = s;
}

// In-place row-wise log_softmax on (B, V). grid.x = B, block = 256.
__global__ __launch_bounds__(256) void k_logsoftmax(float* __restrict__ base) {
    __shared__ float red[256];
    float* row = base + (size_t)blockIdx.x * V_;
    const int tid = threadIdx.x;

    float m = -INFINITY;
    for (int i = tid; i < V_; i += 256) m = fmaxf(m, row[i]);
    red[tid] = m; __syncthreads();
    for (int s = 128; s > 0; s >>= 1) {
        if (tid < s) red[tid] = fmaxf(red[tid], red[tid + s]);
        __syncthreads();
    }
    const float mx = red[0]; __syncthreads();

    float sum = 0.0f;
    for (int i = tid; i < V_; i += 256) sum += __expf(row[i] - mx);
    red[tid] = sum; __syncthreads();
    for (int s = 128; s > 0; s >>= 1) {
        if (tid < s) red[tid] += red[tid + s];
        __syncthreads();
    }
    const float c = mx + logf(red[0]);
    for (int i = tid; i < V_; i += 256) row[i] -= c;
}

// Global flat argmax over (B, V); first occurrence wins. Single block.
__global__ __launch_bounds__(1024) void k_argmax(const float* __restrict__ logp,
                                                 float* __restrict__ out_tok) {
    __shared__ float vred[1024];
    __shared__ int   ired[1024];
    const int tid = threadIdx.x;
    float best = -INFINITY; int bi = 0;
    for (int i = tid; i < B_ * V_; i += 1024) {
        const float v = logp[i];
        if (v > best) { best = v; bi = i; }
    }
    vred[tid] = best; ired[tid] = bi; __syncthreads();
    for (int s = 512; s > 0; s >>= 1) {
        if (tid < s) {
            const float vo = vred[tid + s]; const int io = ired[tid + s];
            if (vo > vred[tid] || (vo == vred[tid] && io < ired[tid])) {
                vred[tid] = vo; ired[tid] = io;
            }
        }
        __syncthreads();
    }
    if (tid == 0) out_tok[0] = (float)ired[0];
}

__global__ void k_cvt_bf16(const float* __restrict__ s, bf16_t* __restrict__ d, int n) {
    for (int i = blockIdx.x * blockDim.x + threadIdx.x; i < n;
         i += gridDim.x * blockDim.x)
        d[i] = (bf16_t)s[i];
}

__global__ void k_fill_f32(float* __restrict__ p, int n, float v) {
    for (int i = blockIdx.x * blockDim.x + threadIdx.x; i < n;
         i += gridDim.x * blockDim.x)
        p[i] = v;
}

__global__ void k_seq_init(float* __restrict__ seq) {
    if (threadIdx.x == 0) seq[0] = (float)BOS_ID;
}

// ---------------------------------------------------------------------------
extern "C" void kernel_launch(void* const* d_in, const int* in_sizes, int n_in,
                              void* d_out, int out_size, void* d_ws, size_t ws_size,
                              hipStream_t stream) {
    (void)in_sizes; (void)n_in; (void)out_size; (void)ws_size;

    const int*   input   = (const int*)  d_in[0];
    const float* eps     = (const float*)d_in[1];
    const float* emb_enc = (const float*)d_in[2];
    const float* Wih_e   = (const float*)d_in[3];
    const float* Whh_e   = (const float*)d_in[4];
    const float* bih_e   = (const float*)d_in[5];
    const float* bhh_e   = (const float*)d_in[6];
    const float* W_enc   = (const float*)d_in[7];
    const float* b_enc   = (const float*)d_in[8];
    const float* W_loc   = (const float*)d_in[9];
    const float* b_loc   = (const float*)d_in[10];
    const float* W_scale = (const float*)d_in[11];
    const float* b_scale = (const float*)d_in[12];
    const float* emb_dec = (const float*)d_in[13];
    const float* W_dec   = (const float*)d_in[14];
    const float* b_dec   = (const float*)d_in[15];
    const float* Wih_d   = (const float*)d_in[16];
    const float* Whh_d   = (const float*)d_in[17];
    const float* bih_d   = (const float*)d_in[18];
    const float* bhh_d   = (const float*)d_in[19];
    const float* W_voc   = (const float*)d_in[20];
    const float* b_voc   = (const float*)d_in[21];

    float* out = (float*)d_out;
    float* out_logp  = out;                                   // (T,B,V)
    float* out_seq   = out + (size_t)T_ * B_ * V_;            // (T+1)
    float* out_loc   = out_seq + (T_ + 1);                    // (B,Z)
    float* out_scale = out_loc + B_ * Z_;                     // (B,Z)

    // -------- workspace bump allocator --------
    char* w = (char*)d_ws;
    size_t off = 0;
    auto alloc = [&](size_t bytes) -> void* {
        void* p = w + off;
        off = (off + bytes + 255) & ~(size_t)255;
        return p;
    };
    bf16_t* emb_enc_bf = (bf16_t*)alloc((size_t)V_ * E_ * 2);
    bf16_t* Wih_e_bf   = (bf16_t*)alloc((size_t)G3H * E_ * 2);
    bf16_t* Whh_e_bf   = (bf16_t*)alloc((size_t)G3H * H_ * 2);
    bf16_t* Whh_d_bf   = (bf16_t*)alloc((size_t)G3H * H_ * 2);
    bf16_t* Wvoc_bf    = (bf16_t*)alloc((size_t)V_ * H_ * 2);
    float*  gx    = (float*)alloc((size_t)S_ * B_ * G3H * 4);
    float*  h     = (float*)alloc((size_t)B_ * H_ * 4);
    bf16_t* h_bf  = (bf16_t*)alloc((size_t)B_ * H_ * 2);
    float*  hg    = (float*)alloc((size_t)B_ * G3H * 4);
    float*  henc  = (float*)alloc((size_t)B_ * Z_ * 4);
    float*  locb  = (float*)alloc((size_t)B_ * Z_ * 4);
    float*  sclb  = (float*)alloc((size_t)B_ * Z_ * 4);
    float*  zvec  = (float*)alloc((size_t)B_ * Z_ * 4);
    float*  gxd   = (float*)alloc((size_t)G3H * 4);

    auto cvt = [&](const float* s, bf16_t* d, int n) {
        int blocks = (n + 255) / 256; if (blocks > 4096) blocks = 4096;
        k_cvt_bf16<<<blocks, 256, 0, stream>>>(s, d, n);
    };
    auto smem_for = [](int K) -> size_t { return (size_t)16 * (K + LDS_PAD) * 2; };

    // -------- one-time weight downconversion (fp32 -> bf16) --------
    cvt(emb_enc, emb_enc_bf, V_ * E_);
    cvt(Wih_e,   Wih_e_bf,   G3H * E_);
    cvt(Whh_e,   Whh_e_bf,   G3H * H_);
    cvt(Whh_d,   Whh_d_bf,   G3H * H_);
    cvt(W_voc,   Wvoc_bf,    V_ * H_);

    // -------- encoder input GEMM: gx = emb[input] @ Wih_e^T + bih_e --------
    {
        dim3 grid(G3H / 128, (S_ * B_) / 16);
        k_wmma_gemm<<<grid, 256, smem_for(E_), stream>>>(
            emb_enc_bf, input, Wih_e_bf, bih_e, gx, G3H, E_);
    }

    // -------- encoder GRU scan --------
    k_fill_f32<<<(B_ * H_ + 255) / 256, 256, 0, stream>>>(h, B_ * H_, 0.0f);
    cvt(h, h_bf, B_ * H_);
    {
        dim3 grid(G3H / 128, B_ / 16);
        const int gate_blocks = (B_ * H_ + 255) / 256;
        for (int s = 0; s < S_; ++s) {
            k_wmma_gemm<<<grid, 256, smem_for(H_), stream>>>(
                h_bf, nullptr, Whh_e_bf, bhh_e, hg, G3H, H_);
            k_gru_gate<<<gate_blocks, 256, 0, stream>>>(
                gx + (size_t)s * B_ * G3H, hg, h, h_bf, 0);
        }
    }

    // -------- latent head + reparametrize --------
    k_dense_f32<<<(B_ * Z_ + 255) / 256, 256, 0, stream>>>(h, W_enc, b_enc, henc, B_, Z_, H_, 0);
    k_dense_f32<<<(B_ * Z_ + 255) / 256, 256, 0, stream>>>(henc, W_loc, b_loc, locb, B_, Z_, Z_, 0);
    k_dense_f32<<<(B_ * Z_ + 255) / 256, 256, 0, stream>>>(henc, W_scale, b_scale, sclb, B_, Z_, Z_, 1);
    k_reparam<<<(B_ * Z_ + 255) / 256, 256, 0, stream>>>(locb, sclb, eps, zvec, out_loc, out_scale);

    // -------- decoder init --------
    k_dense_f32<<<(B_ * H_ + 255) / 256, 256, 0, stream>>>(zvec, W_dec, b_dec, h, B_, H_, Z_, 0);
    cvt(h, h_bf, B_ * H_);
    k_dense_f32<<<(G3H + 255) / 256, 256, 0, stream>>>(
        emb_dec + (size_t)EOS_ID * E_, Wih_d, bih_d, gxd, 1, G3H, E_, 0);
    k_seq_init<<<1, 32, 0, stream>>>(out_seq);

    // -------- decode T steps --------
    {
        dim3 grid_h(G3H / 128, B_ / 16);
        dim3 grid_v(V_ / 128, B_ / 16);
        const int gate_blocks = (B_ * H_ + 255) / 256;
        for (int t = 0; t < T_; ++t) {
            float* logp_t = out_logp + (size_t)t * B_ * V_;
            k_wmma_gemm<<<grid_h, 256, smem_for(H_), stream>>>(
                h_bf, nullptr, Whh_d_bf, bhh_d, hg, G3H, H_);
            k_gru_gate<<<gate_blocks, 256, 0, stream>>>(gxd, hg, h, h_bf, 1);
            k_wmma_gemm<<<grid_v, 256, smem_for(H_), stream>>>(
                h_bf, nullptr, Wvoc_bf, b_voc, logp_t, V_, H_);
            k_logsoftmax<<<B_, 256, 0, stream>>>(logp_t);
            k_argmax<<<1, 1024, 0, stream>>>(logp_t, out_seq + 1 + t);
        }
    }
}
